// DeformableConv2D_54700703482352
// MI455X (gfx1250) — compile-verified, matrix operand order checked by reading, not run on hardware
//
#include <hip/hip_runtime.h>

typedef __attribute__((ext_vector_type(16))) _Float16 v16h;
typedef __attribute__((ext_vector_type(8)))  _Float16 v8h;
typedef __attribute__((ext_vector_type(8)))  float    v8f;

#define Bsz   16
#define Hh    64
#define Ww    64
#define Cc    64
#define K2n   9
#define Ff    128
#define OFFCH 18
#define NPIX  (Bsz * Hh * Ww)   /* 65536 */
#define MT    32                /* pixels (M rows) per block */

// -------------------------------------------------------------------------
// Kernel 0: repack w_conv (576x128 f32, row-major K x N) into f16 WMMA
// B-fragment order: [tap(9)][ktile(2)][ntile(8)][lane(32)][e(16)] halves.
// Fragment mapping (16-bit operand, 32x16 B tile, wave32):
//   lane = n_local + 16*((kk>>3)&1),  e = (kk&7) | ((kk>>4)<<3)
// -------------------------------------------------------------------------
__global__ __launch_bounds__(256) void pack_wconv(const float* __restrict__ w_conv,
                                                  _Float16* __restrict__ wpack) {
  int i = blockIdx.x * 256 + threadIdx.x;
  if (i >= 9 * 2 * 8 * 32 * 16) return;
  int e     = i & 15;
  int lane  = (i >> 4) & 31;
  int rest  = i >> 9;
  int ntile = rest & 7;
  int ktile = (rest >> 3) & 1;
  int tap   = rest >> 4;
  int nloc  = lane & 15;
  int hi    = lane >> 4;
  int kk    = (e & 7) | (hi << 3) | ((e >> 3) << 4);
  int kg    = tap * 64 + ktile * 32 + kk;         // global K index 0..575
  wpack[i]  = (_Float16)w_conv[kg * Ff + ntile * 16 + nloc];
}

// -------------------------------------------------------------------------
// Kernel 1: offset conv (3x3, C=64 -> 18), SAME padding, NHWC.
// w_off staged in LDS (41.5 KB of 320 KB WGP LDS); 18 f32 accumulators.
// -------------------------------------------------------------------------
__global__ __launch_bounds__(256) void offset_conv(const float* __restrict__ x,
                                                   const float* __restrict__ w_off,
                                                   const float* __restrict__ b_off,
                                                   float* __restrict__ offs) {
  __shared__ float sw[K2n * Cc * OFFCH];   // 10368 floats
  for (int j = threadIdx.x; j < K2n * Cc * OFFCH; j += 256) sw[j] = w_off[j];
  __syncthreads();

  int p = blockIdx.x * 256 + threadIdx.x;
  int b = p >> 12;
  int h = (p >> 6) & 63;
  int w = p & 63;

  float acc[OFFCH];
#pragma unroll
  for (int oc = 0; oc < OFFCH; ++oc) acc[oc] = b_off[oc];

  for (int t = 0; t < 9; ++t) {
    int kh = t / 3, kw = t - kh * 3;
    int y = h + kh - 1, xw = w + kw - 1;
    if ((unsigned)y < 64u && (unsigned)xw < 64u) {
      const float* xp = x + (size_t)(((b << 6) + y) * 64 + xw) * Cc;
      const float* wp = sw + t * Cc * OFFCH;
      for (int c = 0; c < Cc; ++c) {
        float xv = xp[c];
        const float* wr = wp + c * OFFCH;
#pragma unroll
        for (int oc = 0; oc < OFFCH; ++oc) acc[oc] = fmaf(xv, wr[oc], acc[oc]);
      }
    }
  }
  float* op = offs + (size_t)p * OFFCH;
#pragma unroll
  for (int oc = 0; oc < OFFCH; ++oc) op[oc] = acc[oc];
}

// -------------------------------------------------------------------------
// Kernel 2: fused bilinear sampling + GEMM via v_wmma_f32_16x16x32_f16.
// One 256-thread block (8 wave32) per 32-pixel row block:
//   M = 32 (two 16-row m-tiles), N = 128 (one 16-wide n-tile per wave),
//   K = 576 chunked by tap (64) = 2 WMMA K-steps of 32.
//   Each wave owns two C tiles (m-tile 0/1) sharing the same B fragments
//   -> 4 WMMAs per tap per wave, B loads amortized 2x.
// A staging is double-buffered in LDS (fragment order) -> ONE barrier/tap:
//   reads of buf[i&1] at iter i and the rewrite at iter i+2 are separated
//   by the iter i+1 barrier.
// -------------------------------------------------------------------------
__global__ __launch_bounds__(256) void deform_gemm(const float* __restrict__ x,
                                                   const float* __restrict__ offs,
                                                   const _Float16* __restrict__ wpack,
                                                   const float* __restrict__ b_conv,
                                                   float* __restrict__ out) {
  __shared__ float    s_w[K2n][MT][4];          // bilinear weights
  __shared__ int      s_idx[K2n][MT][4];        // 4 corner base offsets
  __shared__ _Float16 s_afrag[2][2][2][512];    // [buf][mtile][ktile][lane*16+e]

  const int tid        = threadIdx.x;
  const int pixel_base = blockIdx.x * MT;

  // ---- Phase 1: per (pixel, tap) bilinear weights & corner indices ----
  for (int idx = tid; idx < K2n * MT; idx += 256) {
    int tap = idx >> 5;        // idx / 32
    int m   = idx & 31;
    int p   = pixel_base + m;
    int b   = p >> 12;
    int h   = (p >> 6) & 63;
    int w   = p & 63;
    const float* ob = offs + (size_t)p * OFFCH;
    float xo = ob[2 * tap];
    float yo = ob[2 * tap + 1];
    // replicate reference clip/floor semantics exactly
    float nx = fminf(fmaxf((float)w + xo, 0.0f), 63.0f);
    float ny = fminf(fmaxf((float)h + yo, 0.0f), 63.0f);
    float x0f = floorf(nx), y0f = floorf(ny);
    int x0 = (int)x0f, y0 = (int)y0f;
    int x1 = min(x0 + 1, 63), y1 = min(y0 + 1, 63);
    float ax = (float)x1 - nx, bx = nx - x0f;
    float ay = (float)y1 - ny, by = ny - y0f;
    s_w[tap][m][0] = ax * ay;   // (y0,x0)
    s_w[tap][m][1] = ax * by;   // (y1,x0)
    s_w[tap][m][2] = bx * ay;   // (y0,x1)
    s_w[tap][m][3] = bx * by;   // (y1,x1)
    int base = b << 12;
    s_idx[tap][m][0] = (base + y0 * 64 + x0) << 6;
    s_idx[tap][m][1] = (base + y1 * 64 + x0) << 6;
    s_idx[tap][m][2] = (base + y0 * 64 + x1) << 6;
    s_idx[tap][m][3] = (base + y1 * 64 + x1) << 6;
  }
  __syncthreads();

  const int lane = tid & 31;
  const int wave = tid >> 5;            // n-tile 0..7
  v8f acc0 = {}, acc1 = {};

  // ---- writer role: 8 consecutive channels of one pixel per tap ----
  const int wm  = tid >> 3;             // pixel row 0..31
  const int wc0 = (tid & 7) << 3;       // channel base 0,8,...,56
  const int wmt = wm >> 4;              // writer m-tile
  const int wml = wm & 15;
  const int r0    = wc0 & 31;
  const int kt_w  = wc0 >> 5;                          // writer k-tile
  const int alane = wml + (((r0 >> 3) & 1) << 4);      // fragment lane
  const int e0    = ((r0 >> 4) & 1) << 3;              // fragment elem base

  for (int tap = 0; tap < 9; ++tap) {
    const int buf = tap & 1;
    // ---- build A fragment: sampled[32 x 64] f16, fragment order ----
    float wa = s_w[tap][wm][0], wb = s_w[tap][wm][1];
    float wcc = s_w[tap][wm][2], wd = s_w[tap][wm][3];
    const float* xa = x + s_idx[tap][wm][0] + wc0;
    const float* xb = x + s_idx[tap][wm][1] + wc0;
    const float* xc = x + s_idx[tap][wm][2] + wc0;
    const float* xd = x + s_idx[tap][wm][3] + wc0;
    float4 a0 = *(const float4*)xa, a1 = *(const float4*)(xa + 4);
    float4 b0v = *(const float4*)xb, b1v = *(const float4*)(xb + 4);
    float4 c0v = *(const float4*)xc, c1v = *(const float4*)(xc + 4);
    float4 d0v = *(const float4*)xd, d1v = *(const float4*)(xd + 4);
    float pa[8] = {a0.x, a0.y, a0.z, a0.w, a1.x, a1.y, a1.z, a1.w};
    float pb[8] = {b0v.x, b0v.y, b0v.z, b0v.w, b1v.x, b1v.y, b1v.z, b1v.w};
    float pc[8] = {c0v.x, c0v.y, c0v.z, c0v.w, c1v.x, c1v.y, c1v.z, c1v.w};
    float pd[8] = {d0v.x, d0v.y, d0v.z, d0v.w, d1v.x, d1v.y, d1v.z, d1v.w};
    v8h hv;
#pragma unroll
    for (int j = 0; j < 8; ++j)
      hv[j] = (_Float16)(wa * pa[j] + wb * pb[j] + wcc * pc[j] + wd * pd[j]);
    *(v8h*)&s_afrag[buf][wmt][kt_w][alane * 16 + e0] = hv;   // one ds_store_b128

    __syncthreads();   // single barrier per tap (double-buffered staging)

    // ---- 4 WMMAs: two m-tiles x two K-steps, shared B fragments ----
    v16h fa00 = *(const v16h*)&s_afrag[buf][0][0][lane * 16];
    v16h fa01 = *(const v16h*)&s_afrag[buf][0][1][lane * 16];
    v16h fa10 = *(const v16h*)&s_afrag[buf][1][0][lane * 16];
    v16h fa11 = *(const v16h*)&s_afrag[buf][1][1][lane * 16];
    v16h fb0 = *(const v16h*)(wpack + (((size_t)(tap * 2 + 0) * 8 + wave) * 32 + lane) * 16);
    v16h fb1 = *(const v16h*)(wpack + (((size_t)(tap * 2 + 1) * 8 + wave) * 32 + lane) * 16);
    acc0 = __builtin_amdgcn_wmma_f32_16x16x32_f16(false, fa00, false, fb0,
                                                  (short)0, acc0, false, false);
    acc1 = __builtin_amdgcn_wmma_f32_16x16x32_f16(false, fa10, false, fb0,
                                                  (short)0, acc1, false, false);
    acc0 = __builtin_amdgcn_wmma_f32_16x16x32_f16(false, fa01, false, fb1,
                                                  (short)0, acc0, false, false);
    acc1 = __builtin_amdgcn_wmma_f32_16x16x32_f16(false, fa11, false, fb1,
                                                  (short)0, acc1, false, false);
  }

  // ---- store C fragments + bias: VGPR r -> M = r + 8*(lane>>4), N = lane&15
  int n = (wave << 4) + (lane & 15);
  float bias = b_conv[n];
  int mrow = ((lane >> 4) << 3);
#pragma unroll
  for (int r = 0; r < 8; ++r)
    out[(size_t)(pixel_base + mrow + r) * Ff + n] = acc0[r] + bias;
#pragma unroll
  for (int r = 0; r < 8; ++r)
    out[(size_t)(pixel_base + 16 + mrow + r) * Ff + n] = acc1[r] + bias;
}

// -------------------------------------------------------------------------
extern "C" void kernel_launch(void* const* d_in, const int* in_sizes, int n_in,
                              void* d_out, int out_size, void* d_ws, size_t ws_size,
                              hipStream_t stream) {
  const float* x      = (const float*)d_in[0];
  const float* w_off  = (const float*)d_in[1];
  const float* b_off  = (const float*)d_in[2];
  const float* w_conv = (const float*)d_in[3];
  const float* b_conv = (const float*)d_in[4];
  float* out = (float*)d_out;

  // workspace: [0, 4.72MB) offsets f32, then 147KB packed f16 w_conv
  float*    offs  = (float*)d_ws;
  _Float16* wpack = (_Float16*)((char*)d_ws + (size_t)NPIX * OFFCH * sizeof(float));

  pack_wconv <<<324, 256, 0, stream>>>(w_conv, wpack);          // 73728 elems
  offset_conv<<<NPIX / 256, 256, 0, stream>>>(x, w_off, b_off, offs);
  deform_gemm<<<NPIX / MT, 256, 0, stream>>>(x, offs, wpack, b_conv, out);
}
